// RNN_37185826849416
// MI455X (gfx1250) — matrix operand order repaired and checked by reading, not compiled, not run on hardware
//
#include <hip/hip_runtime.h>

typedef __attribute__((ext_vector_type(16))) _Float16 v16h;
typedef __attribute__((ext_vector_type(8)))  _Float16 v8h;
typedef __attribute__((ext_vector_type(8)))  float    v8f;

#define H_     64
#define T_     512
#define BM     16        // batch rows per block
#define WS     72        // padded row stride in halfs: 144B, 16B-aligned, conflict-free
#define XS     65        // x_s row stride (floats)
#define CHUNK  64        // x staging chunk (timesteps)
#define NW     4         // waves per block (one per hidden-column tile)
#define NT     (NW * 32)

#if defined(__has_builtin)
#if __has_builtin(__builtin_amdgcn_tanhf)
#define HAS_HW_TANH 1
#endif
#endif

__device__ __forceinline__ float tanh_fast(float v) {
#ifdef HAS_HW_TANH
    return __builtin_amdgcn_tanhf(v);      // v_tanh_f32
#else
    float e = __expf(-2.0f * v);
    return (1.0f - e) / (1.0f + e);
#endif
}
__device__ __forceinline__ float sigmoid_fast(float v) {
    return 0.5f * tanh_fast(0.5f * v) + 0.5f;
}

__global__ __launch_bounds__(NT) void lstm_wmma_kernel(
    const float* __restrict__ x,    // [B, T]
    const float* __restrict__ W_ih, // [256]
    const float* __restrict__ W_hh, // [256, 64]
    const float* __restrict__ b_ih, // [256]
    const float* __restrict__ b_hh, // [256]
    const float* __restrict__ W_d,  // [64]
    const float* __restrict__ b_d,  // [1]
    float* __restrict__ out)        // [B]
{
    __shared__ _Float16 Whh_s[256 * WS];    // W_hh[n][k] in f16, padded rows
    __shared__ _Float16 h_s[2][BM * WS];    // ping-pong h state [M][k], f16
    __shared__ float    x_s[BM * XS];       // staged x chunk
    __shared__ float    Wd_s[H_];

    const int tid  = threadIdx.x;
    const int wave = tid >> 5;             // 0..3: owns hidden cols 16*wave..16*wave+15
    const int lane = tid & 31;
    const int base = blockIdx.x * BM;

    // ---- preload weights into LDS (all 128 threads) ----
    for (int i = tid; i < 256 * H_; i += NT) {
        int n = i >> 6, k = i & 63;
        Whh_s[n * WS + k] = (_Float16)W_hh[i];
    }
    if (tid < H_) Wd_s[tid] = W_d[tid];
    for (int i = tid; i < BM * WS; i += NT) h_s[0][i] = (_Float16)0.0f;   // h0 = 0
    __syncthreads();

    const int half_sel = lane >> 4;
    const int lmod     = lane & 15;
    const int jh       = wave;

    // loop-invariant per-lane xproj coefficients for this wave's 4 gate tiles
    float wv[4], bv[4];
    #pragma unroll
    for (int gg = 0; gg < 4; ++gg) {
        int col = 16 * (jh + 4 * gg) + lmod;
        wv[gg] = W_ih[col];
        bv[gg] = b_ih[col] + b_hh[col];
    }

    v8f c = {};   // this wave's c tile (hidden cols 16*jh..16*jh+15, C-layout)

    #pragma unroll 1
    for (int t = 0; t < T_; ++t) {
        const int tc = t & (CHUNK - 1);
        if (tc == 0) {
            for (int i = tid; i < BM * CHUNK; i += NT) {
                int r = i >> 6, cc2 = i & 63;
                x_s[r * XS + cc2] = x[(base + r) * T_ + t + cc2];
            }
            __syncthreads();
        }

        const int p = t & 1;   // read h from buffer p, write new h to 1-p

        // ---- full h (prev step) as f16 A fragments ----
        v16h a0, a1;
        {
            const _Float16* hp = &h_s[p][lmod * WS];
            const int kb = half_sel * 8;
            v8h p0 = *(const v8h*)(hp + kb);
            v8h p1 = *(const v8h*)(hp + kb + 16);
            v8h p2 = *(const v8h*)(hp + kb + 32);
            v8h p3 = *(const v8h*)(hp + kb + 48);
            #pragma unroll
            for (int i = 0; i < 8; ++i) {
                a0[i] = p0[i]; a0[i + 8] = p1[i];
                a1[i] = p2[i]; a1[i + 8] = p3[i];
            }
        }

        // x broadcast for the 8 C-layout rows (M = r + 8*half_sel)
        float xv[8];
        #pragma unroll
        for (int r = 0; r < 8; ++r)
            xv[r] = x_s[(r + 8 * half_sel) * XS + tc];

        // ---- this wave's 4 gate tiles: i, f, g, o for hidden cols 16*jh.. ----
        v8f acc[4];
        #pragma unroll
        for (int gg = 0; gg < 4; ++gg) {
            const int col = 16 * (jh + 4 * gg) + lmod;   // N column of this lane

            v8f a;
            #pragma unroll
            for (int r = 0; r < 8; ++r) a[r] = xv[r] * wv[gg] + bv[gg];  // xproj + bias

            const _Float16* wp = &Whh_s[col * WS + 8 * half_sel];
            v8h b0lo = *(const v8h*)wp;          // K frag0: rows kb..kb+7, 16+kb..16+kb+7
            v8h b0hi = *(const v8h*)(wp + 16);
            v8h b1lo = *(const v8h*)(wp + 32);   // K frag1: rows 32+kb..32+kb+7, 48+kb..48+kb+7
            v8h b1hi = *(const v8h*)(wp + 48);
            v16h bf0, bf1;
            #pragma unroll
            for (int i = 0; i < 8; ++i) {
                bf0[i] = b0lo[i]; bf0[i + 8] = b0hi[i];
                bf1[i] = b1lo[i]; bf1[i + 8] = b1hi[i];
            }
            a = __builtin_amdgcn_wmma_f32_16x16x32_f16(
                    false, a0, false, bf0, (short)0, a, false, false);
            a = __builtin_amdgcn_wmma_f32_16x16x32_f16(
                    false, a1, false, bf1, (short)0, a, false, false);
            acc[gg] = a;
        }

        // elementwise LSTM cell update; write h into the other buffer
        _Float16* hsw = &h_s[1 - p][(8 * half_sel) * WS + 16 * jh + lmod];
        #pragma unroll
        for (int r = 0; r < 8; ++r) {
            float ig = sigmoid_fast(acc[0][r]);
            float fg = sigmoid_fast(acc[1][r]);
            float gt = tanh_fast(acc[2][r]);
            float og = sigmoid_fast(acc[3][r]);
            float cv = fg * c[r] + ig * gt;
            c[r] = cv;
            hsw[r * WS] = (_Float16)(og * tanh_fast(cv));   // M = r + 8*half_sel
        }
        __syncthreads();   // all waves' h stores visible before next step's A loads
    }

    // ---- out[b] = h_last[b] . W_d + b_d  (h_last is in buffer 0 since T is even) ----
    if (tid < BM) {
        float s = b_d[0];
        #pragma unroll
        for (int k = 0; k < H_; ++k)
            s += (float)h_s[0][tid * WS + k] * Wd_s[k];
        out[base + tid] = s;
    }
}

extern "C" void kernel_launch(void* const* d_in, const int* in_sizes, int n_in,
                              void* d_out, int out_size, void* d_ws, size_t ws_size,
                              hipStream_t stream) {
    const float* x    = (const float*)d_in[0];
    const float* W_ih = (const float*)d_in[1];
    const float* W_hh = (const float*)d_in[2];
    const float* b_ih = (const float*)d_in[3];
    const float* b_hh = (const float*)d_in[4];
    const float* W_d  = (const float*)d_in[5];
    const float* b_d  = (const float*)d_in[6];
    float* out = (float*)d_out;

    const int B = in_sizes[0] / T_;   // 2048
    lstm_wmma_kernel<<<B / BM, NT, 0, stream>>>(x, W_ih, W_hh, b_ih, b_hh, W_d, b_d, out);
}
